// StateSpaceModel_82669530513867
// MI455X (gfx1250) — matrix operand order repaired and checked by reading, not compiled, not run on hardware
//
#include <hip/hip_runtime.h>
#include <hip/hip_bf16.h>
#include <stdint.h>
#include <stddef.h>

// Problem constants (from reference): B=32, L=2048, D=512
#define B_  32
#define L_  2048
#define D_  512
#define M_  (B_ * L_)     // 65536 rows for the big GEMMs

// LDS weight panel: 512 rows x 64 cols bf16, padded to 72 elems/row
// (16B of pad per 128B row -> keeps ds_load_b128 alignment, ~2-way banks)
#define PANEL_LD     72
#define PANEL_BYTES  (512 * PANEL_LD * 2)   // 73728 B (dynamic LDS; WGP has 320KB)

typedef __attribute__((ext_vector_type(16))) __bf16 v16bf;
typedef __attribute__((ext_vector_type(8)))  __bf16 v8bf;
typedef __attribute__((ext_vector_type(8)))  float  v8f;
typedef unsigned int ui32x4 __attribute__((ext_vector_type(4)));
typedef int          i32x8  __attribute__((ext_vector_type(8)));
typedef int          i32x4  __attribute__((ext_vector_type(4)));

// ---------------------------------------------------------------------------
// WMMA operand helpers (wave32, v_wmma_f32_16x16x32_bf16)
//
// A operand (16x32 bf16), per CDNA5 ISA table:
//   lane<16 : row M=lane,    elems[0..7]=K 0..7,   elems[8..15]=K 16..23
//   lane>=16: row M=lane-16, elems[0..7]=K 8..15,  elems[8..15]=K 24..31
// B operand (32x16 bf16): lane l holds column N=l&15, K in the same order as A.
// C/D (16x16 f32): lane l, vgpr v -> M = v + (l<16?0:8), N = l&15.
// ---------------------------------------------------------------------------

__device__ __forceinline__ v16bf cat8(v8bf lo, v8bf hi) {
  return __builtin_shufflevector(lo, hi, 0, 1, 2, 3, 4, 5, 6, 7,
                                         8, 9, 10, 11, 12, 13, 14, 15);
}

__device__ __forceinline__ v16bf load_a_bf(const __bf16* p) {
  v8bf lo = *(const v8bf*)(p);
  v8bf hi = *(const v8bf*)(p + 16);
  return cat8(lo, hi);
}

__device__ __forceinline__ v16bf load_a_f32(const float* p) {
  const float4* q = (const float4*)p;
  float4 f0 = q[0], f1 = q[1];   // K base1+0..7
  float4 f2 = q[4], f3 = q[5];   // K base1+16..23
  v16bf r;
  r[0]  = (__bf16)f0.x; r[1]  = (__bf16)f0.y; r[2]  = (__bf16)f0.z; r[3]  = (__bf16)f0.w;
  r[4]  = (__bf16)f1.x; r[5]  = (__bf16)f1.y; r[6]  = (__bf16)f1.z; r[7]  = (__bf16)f1.w;
  r[8]  = (__bf16)f2.x; r[9]  = (__bf16)f2.y; r[10] = (__bf16)f2.z; r[11] = (__bf16)f2.w;
  r[12] = (__bf16)f3.x; r[13] = (__bf16)f3.y; r[14] = (__bf16)f3.z; r[15] = (__bf16)f3.w;
  return r;
}

__device__ __forceinline__ v16bf load_b_col(const __bf16* p, int ld) {
  v16bf r;
#pragma unroll
  for (int s = 0; s < 8; ++s) {
    r[s]     = p[(size_t)s * ld];
    r[8 + s] = p[(size_t)(16 + s) * ld];
  }
  return r;
}
__device__ __forceinline__ v8f zero8() {
  v8f z;
#pragma unroll
  for (int i = 0; i < 8; ++i) z[i] = 0.0f;
  return z;
}

// ---------------------------------------------------------------------------
// Stage a 512(K) x 64(N) bf16 weight panel (row-major, row stride 512) into
// LDS with 16B row padding, via the Tensor Data Mover when available.
// D# per CDNA5 ISA:
//   group0: [1:0]count=1 | lds_addr | global_addr[56:0] | type=2
//   group1: data_size=1(2B), pad_enable, pad_interval=4 (32 DW = one 128B row),
//           pad_amount=3 (4 DW = 16B), tensor_dim0=512, tensor_dim1=512,
//           tile_dim0=64, tile_dim1=512, tensor_dim0_stride=512
//   groups 2/3: zero (2D tile)
// Issued by wave 0 only (TDM ignores EXEC); completion via s_wait_tensorcnt.
// ---------------------------------------------------------------------------
__device__ __forceinline__ void stage_panel(__bf16* sW, const __bf16* gW) {
#if __has_builtin(__builtin_amdgcn_tensor_load_to_lds) && __has_builtin(__builtin_amdgcn_s_wait_tensorcnt)
  if (threadIdx.x < 32) {
    const unsigned long long ga = (unsigned long long)(uintptr_t)gW;
    const unsigned lds_addr = (unsigned)(uintptr_t)sW;  // low 32 bits = LDS byte offset
    ui32x4 g0;
    g0[0] = 1u;                                   // count=1 (valid user descriptor)
    g0[1] = lds_addr;
    g0[2] = (unsigned)(ga & 0xFFFFFFFFu);
    g0[3] = (unsigned)((ga >> 32) & 0x1FFFFFFu) | 0x80000000u;  // type=2 ("image")
    i32x8 g1;
    g1[0] = (int)((1u << 16)      // data_size = 2 bytes
                | (1u << 20)      // pad_enable
                | (4u << 22)      // pad_interval code 4 -> 32 DWORDs (128B)
                | (3u << 25));    // pad_amount  code 3 -> 4 DWORDs (16B)
    g1[1] = (int)(512u << 16);    // tensor_dim0[15:0] = 512 (bits 63:48)
    g1[2] = (int)(512u << 16);    // tensor_dim0[31:16]=0, tensor_dim1[15:0]=512
    g1[3] = (int)(64u << 16);     // tensor_dim1[31:16]=0, tile_dim0=64
    g1[4] = (int)512;             // tile_dim1=512, tile_dim2=0
    g1[5] = (int)512;             // tensor_dim0_stride[31:0] = 512
    g1[6] = 0;                    // tensor_dim0_stride[47:32]=0, dim1_stride lo=0
    g1[7] = 0;                    // tensor_dim1_stride hi = 0
    i32x4 g2 = {0, 0, 0, 0};
    i32x4 g3 = {0, 0, 0, 0};
#if defined(__clang_major__) && (__clang_major__ >= 23)
    i32x8 g4 = {0, 0, 0, 0, 0, 0, 0, 0};
    __builtin_amdgcn_tensor_load_to_lds(g0, g1, g2, g3, g4, 0);
#else
    __builtin_amdgcn_tensor_load_to_lds(g0, g1, g2, g3, 0);
#endif
    __builtin_amdgcn_s_wait_tensorcnt(0);
  }
  __syncthreads();
#else
  // Fallback: cooperative copy with padding (256 threads, 16B chunks)
  for (int i = threadIdx.x; i < 512 * 4; i += 256) {
    const int row = i >> 2, c = (i & 3) * 8;
    *(v8bf*)(sW + (size_t)row * PANEL_LD + c) = *(const v8bf*)(gW + (size_t)row * D_ + c);
  }
  __syncthreads();
#endif
}

// ---------------------------------------------------------------------------
// f32 -> bf16 conversion of the small weight matrices
// ---------------------------------------------------------------------------
__global__ void k_f32_to_bf16(const float* __restrict__ src,
                              __bf16* __restrict__ dst, int n) {
  int i = blockIdx.x * blockDim.x + threadIdx.x;
  if (i < n) dst[i] = (__bf16)src[i];
}

// Zero the h ping-pong exchange buffers and the spin-barrier counters
// (must run every launch: the graph is replayed and counters are monotonic).
__global__ void k_init(__bf16* __restrict__ hx, unsigned* __restrict__ bar) {
  int i = blockIdx.x * blockDim.x + threadIdx.x;
  if (i < 2 * 2 * 16 * D_) hx[i] = (__bf16)0.0f;
  if (i < 2) bar[i] = 0u;
}

// ---------------------------------------------------------------------------
// GEMM 1: U[M,512] = X[M,512](f32) @ W[512,512](bf16), U stored bf16.
// 256 threads = 8 waves; block tile 128(M) x 64(N); wave tile 16x64.
// Weight panel staged once into LDS by the TDM; A tiles double-buffered in registers.
// ---------------------------------------------------------------------------
__global__ __launch_bounds__(256)
void k_gemm_xB(const float* __restrict__ X, const __bf16* __restrict__ W,
               __bf16* __restrict__ U) {
  extern __shared__ __bf16 sW[];
  const int lane  = threadIdx.x & 31;
  const int wave  = threadIdx.x >> 5;
  const int row0  = blockIdx.x * 128 + wave * 16;
  const int col0  = blockIdx.y * 64;
  const int r     = lane & 15;
  const int base1 = (lane < 16) ? 0 : 8;
  const int mbase = (lane < 16) ? 0 : 8;
  const int n     = lane & 15;

  stage_panel(sW, W + col0);   // K=512 x N=64 panel for this block

  v8f acc[4];
#pragma unroll
  for (int t = 0; t < 4; ++t) acc[t] = zero8();

  const float* xrow = X + (size_t)(row0 + r) * D_ + base1;
  const __bf16* brow = sW + (size_t)base1 * PANEL_LD + n;   // + kk*32*PANEL_LD

  v16bf a_cur = load_a_f32(xrow);
#pragma unroll 4
  for (int kk = 0; kk < 16; ++kk) {
    const __bf16* bk = brow + (size_t)kk * 32 * PANEL_LD;
    v16bf b0 = load_b_col(bk, PANEL_LD);
    v16bf b1 = load_b_col(bk + 16, PANEL_LD);
    v16bf b2 = load_b_col(bk + 32, PANEL_LD);
    v16bf b3 = load_b_col(bk + 48, PANEL_LD);
    v16bf a_next = a_cur;
    if (kk + 1 < 16) {
      __builtin_prefetch(xrow + (kk + 2) * 32, 0, 3);
      a_next = load_a_f32(xrow + (kk + 1) * 32);
    }
    acc[0] = __builtin_amdgcn_wmma_f32_16x16x32_bf16(false, a_cur, false, b0, (short)0, acc[0], false, false);
    acc[1] = __builtin_amdgcn_wmma_f32_16x16x32_bf16(false, a_cur, false, b1, (short)0, acc[1], false, false);
    acc[2] = __builtin_amdgcn_wmma_f32_16x16x32_bf16(false, a_cur, false, b2, (short)0, acc[2], false, false);
    acc[3] = __builtin_amdgcn_wmma_f32_16x16x32_bf16(false, a_cur, false, b3, (short)0, acc[3], false, false);
    a_cur = a_next;
  }

#pragma unroll
  for (int t = 0; t < 4; ++t) {
#pragma unroll
    for (int v = 0; v < 8; ++v) {
      U[(size_t)(row0 + mbase + v) * D_ + col0 + t * 16 + n] = (__bf16)acc[t][v];
    }
  }
}

// ---------------------------------------------------------------------------
// GEMM 2: Y[M,512](f32) = HS[M,512](bf16) @ C[512,512](bf16)
// ---------------------------------------------------------------------------
__global__ __launch_bounds__(256)
void k_gemm_hC(const __bf16* __restrict__ HS, const __bf16* __restrict__ W,
               float* __restrict__ Y) {
  extern __shared__ __bf16 sW[];
  const int lane  = threadIdx.x & 31;
  const int wave  = threadIdx.x >> 5;
  const int row0  = blockIdx.x * 128 + wave * 16;
  const int col0  = blockIdx.y * 64;
  const int r     = lane & 15;
  const int base1 = (lane < 16) ? 0 : 8;
  const int mbase = (lane < 16) ? 0 : 8;
  const int n     = lane & 15;

  stage_panel(sW, W + col0);

  v8f acc[4];
#pragma unroll
  for (int t = 0; t < 4; ++t) acc[t] = zero8();

  const __bf16* hrow = HS + (size_t)(row0 + r) * D_ + base1;
  const __bf16* brow = sW + (size_t)base1 * PANEL_LD + n;

  v16bf a_cur = load_a_bf(hrow);
#pragma unroll 4
  for (int kk = 0; kk < 16; ++kk) {
    const __bf16* bk = brow + (size_t)kk * 32 * PANEL_LD;
    v16bf b0 = load_b_col(bk, PANEL_LD);
    v16bf b1 = load_b_col(bk + 16, PANEL_LD);
    v16bf b2 = load_b_col(bk + 32, PANEL_LD);
    v16bf b3 = load_b_col(bk + 48, PANEL_LD);
    v16bf a_next = a_cur;
    if (kk + 1 < 16) {
      __builtin_prefetch(hrow + (kk + 2) * 32, 0, 3);
      a_next = load_a_bf(hrow + (kk + 1) * 32);
    }
    acc[0] = __builtin_amdgcn_wmma_f32_16x16x32_bf16(false, a_cur, false, b0, (short)0, acc[0], false, false);
    acc[1] = __builtin_amdgcn_wmma_f32_16x16x32_bf16(false, a_cur, false, b1, (short)0, acc[1], false, false);
    acc[2] = __builtin_amdgcn_wmma_f32_16x16x32_bf16(false, a_cur, false, b2, (short)0, acc[2], false, false);
    acc[3] = __builtin_amdgcn_wmma_f32_16x16x32_bf16(false, a_cur, false, b3, (short)0, acc[3], false, false);
    a_cur = a_next;
  }

#pragma unroll
  for (int t = 0; t < 4; ++t) {
#pragma unroll
    for (int v = 0; v < 8; ++v) {
      Y[(size_t)(row0 + mbase + v) * D_ + col0 + t * 16 + n] = acc[t][v];
    }
  }
}

// ---------------------------------------------------------------------------
// Scan kernel: h_t = h_{t-1} @ A + u_t for two batch-groups of 16 rows.
// Grid = 32 WGs (group g = blk>>4, N-slice j = blk&15), 64 threads (2 waves).
// Each wave keeps its 512x16 bf16 slice of deltaA resident in registers
// (16 x v16bf = 128 VGPRs); per step: seed accumulator with u_t, chain 16
// WMMAs over K=512, publish h_t bf16 through an L2 ping-pong buffer, sync
// the group's 16 WGs with an agent-scope atomic counter barrier.
// ---------------------------------------------------------------------------
__global__ __launch_bounds__(64)
void k_scan(const __bf16* __restrict__ Abf, const __bf16* __restrict__ U,
            __bf16* __restrict__ HS, __bf16* __restrict__ hx,
            unsigned* __restrict__ bar) {
  const int lane  = threadIdx.x & 31;
  const int wave  = threadIdx.x >> 5;
  const int j     = blockIdx.x & 15;   // N-slice (32 columns)
  const int g     = blockIdx.x >> 4;   // batch group (16 rows)
  const int col0  = j * 32 + wave * 16;
  const int base1 = (lane < 16) ? 0 : 8;
  const int mbase = (lane < 16) ? 0 : 8;
  const int n     = lane & 15;
  const int r     = lane & 15;

  // Resident B operand: deltaA[:, col0:col0+16] as 16 K-tiles of 32.
  v16bf breg[16];
#pragma unroll
  for (int kk = 0; kk < 16; ++kk)
    breg[kk] = load_b_col(Abf + (size_t)(kk * 32 + base1) * D_ + col0 + n, D_);

  __bf16* hx0 = hx + (size_t)g * 2 * 16 * D_;  // ping-pong pair for this group
  const size_t urow0 = (size_t)(g * 16 + mbase) * L_ * D_ + col0 + n;

  for (int t = 0; t < L_; ++t) {
    const __bf16* hr = hx0 + (size_t)(t & 1) * 16 * D_;        // h_{t-1}
    __bf16*       hw = hx0 + (size_t)((t + 1) & 1) * 16 * D_;  // h_t

    // acc = u_t tile (bf16 -> f32)
    v8f acc;
#pragma unroll
    for (int v = 0; v < 8; ++v)
      acc[v] = (float)U[urow0 + (size_t)v * L_ * D_ + (size_t)t * D_];

    // acc += h_{t-1}[16,512] @ A_slice[512,16]
#pragma unroll
    for (int kk = 0; kk < 16; ++kk) {
      v16bf a = load_a_bf(hr + (size_t)r * D_ + kk * 32 + base1);
      acc = __builtin_amdgcn_wmma_f32_16x16x32_bf16(
          false, a, false, breg[kk], (short)0, acc, false, false);
    }

    // Publish h_t: exchange buffer + history for the output GEMM.
#pragma unroll
    for (int v = 0; v < 8; ++v) {
      __bf16 hv = (__bf16)acc[v];
      hw[(size_t)(mbase + v) * D_ + col0 + n] = hv;
      HS[((size_t)(g * 16 + mbase + v) * L_ + t) * D_ + col0 + n] = hv;
    }

    // Inter-workgroup barrier for this group's 16 WGs (sense = step count).
    __threadfence();
    __syncthreads();
    if (threadIdx.x == 0) {
      __hip_atomic_fetch_add(&bar[g], 1u, __ATOMIC_RELEASE,
                             __HIP_MEMORY_SCOPE_AGENT);
      const unsigned target = 16u * (unsigned)(t + 1);
      while (__hip_atomic_load(&bar[g], __ATOMIC_ACQUIRE,
                               __HIP_MEMORY_SCOPE_AGENT) < target) {
        __builtin_amdgcn_s_sleep(1);
      }
    }
    __syncthreads();
    __builtin_amdgcn_fence(__ATOMIC_ACQUIRE, "agent");
  }
}

// ---------------------------------------------------------------------------
// Workspace layout (bytes):
//   [0,            67108864)  U   : u = x@deltaB, bf16 [B*L, D]
//   [67108864,    134217728)  HS  : h history,    bf16 [B*L, D]
//   [134217728,   134742016)  Abf : deltaA bf16
//   [134742016,   135266304)  Bbf : deltaB bf16
//   [135266304,   135790592)  Cbf : C bf16
//   [135790592,   135856128)  HX  : 2 groups x 2 bufs x 16 x 512 bf16
//   [135856128,   135856136)  bar : 2 x u32 barrier counters
// ---------------------------------------------------------------------------
extern "C" void kernel_launch(void* const* d_in, const int* in_sizes, int n_in,
                              void* d_out, int out_size, void* d_ws, size_t ws_size,
                              hipStream_t stream) {
  const float* x  = (const float*)d_in[0];
  const float* dA = (const float*)d_in[1];
  const float* dB = (const float*)d_in[2];
  const float* Cm = (const float*)d_in[3];
  float* y = (float*)d_out;

  char* ws = (char*)d_ws;
  __bf16*   U   = (__bf16*)(ws);
  __bf16*   HS  = (__bf16*)(ws + 67108864ull);
  __bf16*   Abf = (__bf16*)(ws + 134217728ull);
  __bf16*   Bbf = (__bf16*)(ws + 134742016ull);
  __bf16*   Cbf = (__bf16*)(ws + 135266304ull);
  __bf16*   HX  = (__bf16*)(ws + 135790592ull);
  unsigned* bar = (unsigned*)(ws + 135856128ull);

  k_f32_to_bf16<<<dim3(1024), dim3(256), 0, stream>>>(dA, Abf, D_ * D_);
  k_f32_to_bf16<<<dim3(1024), dim3(256), 0, stream>>>(dB, Bbf, D_ * D_);
  k_f32_to_bf16<<<dim3(1024), dim3(256), 0, stream>>>(Cm, Cbf, D_ * D_);

  k_init<<<dim3(128), dim3(256), 0, stream>>>(HX, bar);

  // u = x @ deltaB (TDM-staged weight panel in dynamic LDS)
  k_gemm_xB<<<dim3(M_ / 128, D_ / 64), dim3(256), PANEL_BYTES, stream>>>(x, Bbf, U);

  // Sequential scan: 32 co-resident WGs (2 groups x 16 slices).
  k_scan<<<dim3(32), dim3(64), 0, stream>>>(Abf, U, HS, HX, bar);

  // y = h @ C
  k_gemm_hC<<<dim3(M_ / 128, D_ / 64), dim3(256), PANEL_BYTES, stream>>>(HS, Cbf, y);
}
